// LDR_42339787604900
// MI455X (gfx1250) — matrix operand order repaired and checked by reading, not compiled, not run on hardware
//
#include <hip/hip_runtime.h>

// ---------------- problem constants ----------------
constexpr int kN    = 4096;
constexpr int kB    = 32;
constexpr int kCin  = 4, kCout = 4, kR = 4;
constexpr int kBN   = kB * kN;            // 131072 elements
constexpr int kOut  = kCout * kB * kN;    // 524288 elements
constexpr int kNIJS = kCin * kCout * kR;  // 64
constexpr int kGRP  = 8;                  // (i,s) split for step 2 (2 terms each)

// LDS: 8 element-shifted bf16 copies of the length-2N extended generator
constexpr int STRIDE_E     = 2 * kN + 8;        // 8200 elements per copy (covers q in [0,2N+8))
constexpr int STRIDE_BYTES = STRIDE_E * 2;      // 16400 B, multiple of 16
constexpr int NCOPIES      = 8;
constexpr int SMEM_BYTES   = NCOPIES * STRIDE_BYTES; // 131200 B (dynamic LDS; 320KB/WGP on CDNA5)
constexpr int CHUNKS       = STRIDE_E / 8;      // 1025 aligned 8-element chunks per copy

typedef __attribute__((ext_vector_type(16))) __bf16 v16bf;
typedef __attribute__((ext_vector_type(8)))  float  v8f;

struct alignas(16) U4 { unsigned int x, y, z, w; };
struct alignas(16) F4 { float x, y, z, w; };
union  AB { U4 u[2]; v16bf v; };   // 32B: one WMMA A/B operand (16 bf16 per lane)

__device__ inline unsigned short f2bf(float f) {        // fp32 -> bf16 RNE
  unsigned int x = __float_as_uint(f);
  x += 0x7fffu + ((x >> 16) & 1u);
  return (unsigned short)(x >> 16);
}
__device__ inline unsigned int pk2(float a, float b) {
  return (unsigned int)f2bf(a) | ((unsigned int)f2bf(b) << 16);
}
__device__ inline v8f vzero() {
  v8f z;
  #pragma unroll
  for (int e = 0; e < 8; ++e) z[e] = 0.0f;
  return z;
}

// Build the 8 shifted LDS copies: copy_p[q] = ext[q - p], 16B-aligned b128 stores.
// MODE 0 (step1, H): ext[d] = (d<N) ? -h[d]     : h[d-N]      (C_{-1}(h)^T, negacyclic corr)
// MODE 1 (step2, G): ext[d] = (d<N) ? g[N-1-d]  : g[2N-1-d]   (C_1(g), reversed for ascending reads)
template<int MODE>
__device__ inline void build_copies(char* sm, const float* __restrict__ src) {
  for (int t = threadIdx.x; t < NCOPIES * CHUNKS; t += blockDim.x) {
    const int p  = t / CHUNKS;
    const int q0 = (t - p * CHUNKS) * 8;
    float vf[8];
    #pragma unroll
    for (int e = 0; e < 8; ++e) {
      int d = q0 - p + e;                          // edge chunks clamp; clamped slots never read
      d = d < 0 ? 0 : (d > 2 * kN - 1 ? 2 * kN - 1 : d);
      vf[e] = (MODE == 0) ? ((d < kN) ? -src[d] : src[d - kN])
                          : ((d < kN) ? src[kN - 1 - d] : src[2 * kN - 1 - d]);
    }
    U4 u;
    u.x = pk2(vf[0], vf[1]); u.y = pk2(vf[2], vf[3]);
    u.z = pk2(vf[4], vf[5]); u.w = pk2(vf[6], vf[7]);
    *(U4*)(sm + p * STRIDE_BYTES + q0 * 2) = u;
  }
}

// Toeplitz GEMM core: for RT 16-row k-tiles starting at k0w, accumulate over full K=N:
//   acc[a][h] (16x16, rows k, cols b=16h..16h+15) += A_toep(16x32) x B(32x16)
// A element (row k, col m) = ext[m - k + (N - NCOFF)]; read from shifted copy p=(k+NCOFF)&7
// so every ds_load_b128 is 16B aligned; adjacent row-tiles overlap so the compiler
// turns the A fetches into a sliding window of reused ds_load_b128s.
// B source layout: bf16 [b][m], m contiguous.
template<int RT, int NCOFF>
__device__ inline void toep_gemm(const char* sm, const unsigned short* __restrict__ srcB,
                                 int k0w, v8f (&acc)[RT][2]) {
  const int lane = threadIdx.x & 31;
  const int r  = lane & 15;       // A row within tile / B column (b)
  const int kh = lane >> 4;       // K-half select per WMMA operand layout
  for (int m0 = 0; m0 < kN; m0 += 32) {
    AB b0, b1;
    const U4* pb0 = (const U4*)(srcB + (r     ) * kN + m0 + 16 * kh);
    const U4* pb1 = (const U4*)(srcB + (r + 16) * kN + m0 + 16 * kh);
    b0.u[0] = pb0[0]; b0.u[1] = pb0[1];          // b columns 0..15
    b1.u[0] = pb1[0]; b1.u[1] = pb1[1];          // b columns 16..31
    #pragma unroll
    for (int a = 0; a < RT; ++a) {
      const int k  = k0w + a * 16 + r;
      const int p  = (k + NCOFF) & 7;            // shifted copy -> aligned base
      const int q1 = m0 + (kN - NCOFF) - k + p + 16 * kh;  // q1 % 8 == 0
      const U4* pa = (const U4*)(sm + p * STRIDE_BYTES + q1 * 2);
      AB av; av.u[0] = pa[0]; av.u[1] = pa[1];   // K = c0..c0+15
      acc[a][0] = __builtin_amdgcn_wmma_f32_16x16x32_bf16(false, av.v, false, b0.v,
                                                          (short)0, acc[a][0], false, false);
      acc[a][1] = __builtin_amdgcn_wmma_f32_16x16x32_bf16(false, av.v, false, b1.v,
                                                          (short)0, acc[a][1], false, false);
    }
  }
}

// ---------------- kernels ----------------

__global__ __launch_bounds__(256) void k_cvt_x(const float* __restrict__ x,
                                               unsigned short* __restrict__ x_bf) {
  const int e = blockIdx.x * 256 + threadIdx.x;
  x_bf[e] = f2bf(x[e]);
}

// Step 1: t[ijs] = C_{-1}(H_ijs)^T x[i]   (bf16 out, layout [ijs][b][k])
__global__ __launch_bounds__(256) void k_step1(const float* __restrict__ Hsrc,
                                               const unsigned short* __restrict__ x_bf,
                                               unsigned short* __restrict__ t_bf) {
  extern __shared__ char sm[];
  const int ktile = blockIdx.x;                  // 0..7, 512 k-rows each
  const int ijs   = blockIdx.y;                  // 0..63
  const int i     = ijs >> 4;
  build_copies<0>(sm, Hsrc + ijs * kN);
  __syncthreads();
  const int wav = threadIdx.x >> 5;
  const int k0w = ktile * 512 + wav * 64;        // 8 waves x 64 rows
  v8f acc[4][2];
  #pragma unroll
  for (int a = 0; a < 4; ++a) { acc[a][0] = vzero(); acc[a][1] = vzero(); }
  toep_gemm<4, 0>(sm, x_bf + i * kBN, k0w, acc);
  unsigned short* tb = t_bf + ijs * kBN;
  const int lane = threadIdx.x & 31;
  const int n = lane & 15, kh = lane >> 4;
  #pragma unroll
  for (int a = 0; a < 4; ++a)
    #pragma unroll
    for (int h = 0; h < 2; ++h) {
      const int b   = h * 16 + n;
      const int kst = k0w + a * 16 + 8 * kh;     // 8 consecutive k per lane (D-layout rows)
      U4 u;
      u.x = pk2(acc[a][h][0], acc[a][h][1]);
      u.y = pk2(acc[a][h][2], acc[a][h][3]);
      u.z = pk2(acc[a][h][4], acc[a][h][5]);
      u.w = pk2(acc[a][h][6], acc[a][h][7]);
      *(U4*)(tb + b * kN + kst) = u;
    }
}

// Step 2: partial[grp][j] = sum over 2 of the 16 (i,s) of C_1(G_ijs) t[ijs]
// RT=4 so the steady state matches step 1: 8 WMMA per 4 global_load_b128.
__global__ __launch_bounds__(256) void k_step2(const float* __restrict__ Gsrc,
                                               const unsigned short* __restrict__ t_bf,
                                               float* __restrict__ part) {
  extern __shared__ char sm[];
  const int ktile = blockIdx.x;                  // 0..7, 512 k-rows each
  const int j     = blockIdx.y;                  // 0..3
  const int grp   = blockIdx.z;                  // 0..7 ((i,s) eighth)
  const int wav = threadIdx.x >> 5;
  const int k0w = ktile * 512 + wav * 64;        // 8 waves x 64 rows
  v8f acc[4][2];
  #pragma unroll
  for (int a = 0; a < 4; ++a) { acc[a][0] = vzero(); acc[a][1] = vzero(); }
  for (int isg = 0; isg < 2; ++isg) {
    const int is  = grp * 2 + isg;
    const int ijs = (is >> 2) * 16 + j * 4 + (is & 3);
    __syncthreads();                             // previous GEMM done reading LDS
    build_copies<1>(sm, Gsrc + ijs * kN);
    __syncthreads();
    toep_gemm<4, 1>(sm, t_bf + ijs * kBN, k0w, acc);
  }
  float* pp = part + grp * kOut + j * kBN;
  const int lane = threadIdx.x & 31;
  const int n = lane & 15, kh = lane >> 4;
  #pragma unroll
  for (int a = 0; a < 4; ++a)
    #pragma unroll
    for (int h = 0; h < 2; ++h) {
      const int b   = h * 16 + n;
      const int kst = k0w + a * 16 + 8 * kh;
      F4 lo, hi;
      lo.x = acc[a][h][0]; lo.y = acc[a][h][1]; lo.z = acc[a][h][2]; lo.w = acc[a][h][3];
      hi.x = acc[a][h][4]; hi.y = acc[a][h][5]; hi.z = acc[a][h][6]; hi.w = acc[a][h][7];
      *(F4*)(pp + b * kN + kst)     = lo;
      *(F4*)(pp + b * kN + kst + 4) = hi;
    }
}

// Deterministic 8-way reduction of the (i,s)-group partials.
__global__ __launch_bounds__(256) void k_reduce(const float* __restrict__ part,
                                                float* __restrict__ out) {
  const int e = blockIdx.x * 256 + threadIdx.x;
  float s0 = part[e]            + part[e + kOut];
  float s1 = part[e + 2 * kOut] + part[e + 3 * kOut];
  float s2 = part[e + 4 * kOut] + part[e + 5 * kOut];
  float s3 = part[e + 6 * kOut] + part[e + 7 * kOut];
  out[e] = (s0 + s1) + (s2 + s3);
}

// ---------------- launcher ----------------
extern "C" void kernel_launch(void* const* d_in, const int* in_sizes, int n_in,
                              void* d_out, int out_size, void* d_ws, size_t ws_size,
                              hipStream_t stream) {
  (void)in_sizes; (void)n_in; (void)out_size; (void)ws_size;
  const float* x = (const float*)d_in[0];
  const float* G = (const float*)d_in[1];
  const float* H = (const float*)d_in[2];
  float* out = (float*)d_out;

  char* ws = (char*)d_ws;                           // ~33.5 MB total
  unsigned short* x_bf = (unsigned short*)ws;                         // 1 MB
  unsigned short* t_bf = (unsigned short*)(ws + (size_t)kCin * kBN * 2);          // 16 MB
  float* part = (float*)(ws + (size_t)kCin * kBN * 2 + (size_t)kNIJS * kBN * 2);  // 16 MB

  hipFuncSetAttribute(reinterpret_cast<const void*>(k_step1),
                      hipFuncAttributeMaxDynamicSharedMemorySize, SMEM_BYTES);
  hipFuncSetAttribute(reinterpret_cast<const void*>(k_step2),
                      hipFuncAttributeMaxDynamicSharedMemorySize, SMEM_BYTES);

  k_cvt_x <<<(kCin * kBN) / 256, 256, 0, stream>>>(x, x_bf);
  k_step1 <<<dim3(8, kNIJS), 256, SMEM_BYTES, stream>>>(H, x_bf, t_bf);
  k_step2 <<<dim3(8, kCout, kGRP), 256, SMEM_BYTES, stream>>>(G, t_bf, part);
  k_reduce<<<kOut / 256, 256, 0, stream>>>(part, out);
}
